// mru_lm_block_12232066859477
// MI455X (gfx1250) — matrix operand order repaired and checked by reading, not compiled, not run on hardware
//
#include <hip/hip_runtime.h>
#include <hip/hip_bf16.h>

// ---------------- problem constants ----------------
constexpr int B_ = 4, S_ = 2048, E_ = 2048, H_ = 8, O_ = 16, C_ = 16;
constexpr int M_ = B_ * S_;            // 8192 rows of token activations
constexpr int FF_ = 4 * E_;            // 8192
constexpr int LC_ = 64;                // scan chunk length
constexpr int NCH_ = S_ / LC_;         // 32 chunks per sequence
constexpr float SCALE_ = 0.08f / 16.0f;
constexpr float EPS_ = 1e-5f;

typedef __attribute__((ext_vector_type(16))) __bf16 v16bf;
typedef __attribute__((ext_vector_type(8)))  float  v8f;
typedef __attribute__((ext_vector_type(2)))  float  v2f;

#if defined(__has_builtin)
#if __has_builtin(__builtin_amdgcn_wmma_f32_16x16x4_f32)
#define HAVE_WMMA_F32X4 1
#endif
#endif

// ---------------------------------------------------------------------------
// 16x16 fp32 matmul on one wave32: D = A(16x16) * B(16x16), via 4x
// v_wmma_f32_16x16x4_f32.  A and B are row-major 16x16 float arrays (LDS or
// global).  fp32 A 16x4 layout: lane L holds M=L%16, VGPRv holds K = kb + v +
// 2*(L>=16).  fp32 B 4x16 layout: VGPRv holds row K = kb + v + 2*(L>=16),
// N = L%16 (rows striped across lanes within a VGPR).
// ---------------------------------------------------------------------------
__device__ inline v8f mat16_mul(const float* A, const float* Bm) {
  const int lane = threadIdx.x & 31;
  const int half = lane >> 4;
  const int mq   = lane & 15;
  v8f acc = {};
#if HAVE_WMMA_F32X4
#pragma unroll
  for (int kb = 0; kb < 16; kb += 4) {
    v2f a, b;
#pragma unroll
    for (int v = 0; v < 2; ++v) {
      const int k = kb + v + 2 * half;
      a[v] = A[mq * 16 + k];
      b[v] = Bm[k * 16 + mq];
    }
    acc = __builtin_amdgcn_wmma_f32_16x16x4_f32(false, a, false, b,
                                                (short)0, acc, false, false);
  }
#else
#pragma unroll
  for (int r = 0; r < 8; ++r) {
    float s = 0.f;
    const int mrow = r + 8 * half;
#pragma unroll
    for (int k = 0; k < 16; ++k) s += A[mrow * 16 + k] * Bm[k * 16 + mq];
    acc[r] = s;
  }
#endif
  return acc;
}

// Scatter a 16x16 f32 C/D fragment (VGPRr: M=r / r+8, N=lane%16) row-major.
__device__ inline void store_d16(float* dst, v8f d) {
  const int lane = threadIdx.x & 31;
  const int half = lane >> 4;
  const int n    = lane & 15;
#pragma unroll
  for (int r = 0; r < 8; ++r) dst[(r + 8 * half) * 16 + n] = d[r];
}

// ---------------------------------------------------------------------------
// fp32 -> bf16 conversion
// ---------------------------------------------------------------------------
__global__ __launch_bounds__(256) void cvt_bf16_kernel(const float* __restrict__ in,
                                                       __bf16* __restrict__ out, int n) {
  int i = blockIdx.x * 256 + threadIdx.x;
  if (i < n) out[i] = (__bf16)in[i];
}

// ---------------------------------------------------------------------------
// LayerNorm (weight, no bias) -> bf16, one 256-thread block per row of E=2048.
// ---------------------------------------------------------------------------
__global__ __launch_bounds__(256) void ln_bf16_kernel(const float* __restrict__ x,
                                                      const float* __restrict__ g,
                                                      __bf16* __restrict__ out) {
  const int row = blockIdx.x;
  const int tid = threadIdx.x;
  const float* xr = x + (size_t)row * E_;
  float s1 = 0.f, s2 = 0.f;
#pragma unroll
  for (int j = tid; j < E_; j += 256) { float v = xr[j]; s1 += v; s2 += v * v; }
  __shared__ float r1[256], r2[256];
  r1[tid] = s1; r2[tid] = s2; __syncthreads();
  for (int st = 128; st > 0; st >>= 1) {
    if (tid < st) { r1[tid] += r1[tid + st]; r2[tid] += r2[tid + st]; }
    __syncthreads();
  }
  const float mean = r1[0] * (1.0f / E_);
  const float var  = r2[0] * (1.0f / E_) - mean * mean;
  const float rstd = rsqrtf(var + EPS_);
#pragma unroll
  for (int j = tid; j < E_; j += 256)
    out[(size_t)row * E_ + j] = (__bf16)((xr[j] - mean) * rstd * g[j]);
}

// ---------------------------------------------------------------------------
// bf16 WMMA GEMM: out[M,N] = A[M,K] * Bw[K,N] (+ epilogue)
//   EPI 0: fp32 out = v*SCALE_ + I(o==q)            (builds new_m)
//   EPI 1: fp32 out = v + res[idx]                   (residual / accumulate)
//   EPI 2: bf16 out = gelu_exact(v)
// 128x128 tile / block, 256 threads = 8 waves (4 in M x 2 in N), each wave a
// 32x64 sub-tile => 2x4 v_wmma_f32_16x16x32_bf16 per K-step of 32.
// ---------------------------------------------------------------------------
template <int EPI>
__global__ __launch_bounds__(256)
void gemm_bf16_kernel(const __bf16* __restrict__ Ag, const __bf16* __restrict__ Bg,
                      float* __restrict__ outF, __bf16* __restrict__ outB,
                      const float* __restrict__ res, int Ndim, int Kdim) {
  __shared__ __align__(16) __bf16 As[128][40];   // 32 + 8 pad
  __shared__ __align__(16) __bf16 Bs[32][136];   // 128 + 8 pad

  const int tid  = threadIdx.x;
  const int lane = tid & 31;
  const int half = lane >> 4;
  const int ln   = lane & 15;
  const int wave = tid >> 5;
  const int wm   = wave >> 1;     // 0..3  (32 rows each)
  const int wn   = wave & 1;      // 0..1  (64 cols each)
  const int bM   = blockIdx.y * 128;
  const int bN   = blockIdx.x * 128;

  v8f acc[2][4];
#pragma unroll
  for (int mi = 0; mi < 2; ++mi)
#pragma unroll
    for (int ni = 0; ni < 4; ++ni) acc[mi][ni] = (v8f){};

  for (int k0 = 0; k0 < Kdim; k0 += 32) {
    // ---- stage 128x32 A and 32x128 B slabs into LDS (16B vector ops) ----
#pragma unroll
    for (int i = 0; i < 2; ++i) {
      const int c   = tid + i * 256;          // 512 chunks of 8 bf16 each
      const int ar  = c >> 2, ac = (c & 3) * 8;
      const uint4 av = *reinterpret_cast<const uint4*>(
          Ag + (size_t)(bM + ar) * Kdim + k0 + ac);
      *reinterpret_cast<uint4*>(&As[ar][ac]) = av;
      const int br  = c >> 4, bc = (c & 15) * 8;
      const uint4 bv = *reinterpret_cast<const uint4*>(
          Bg + (size_t)(k0 + br) * Ndim + bN + bc);
      *reinterpret_cast<uint4*>(&Bs[br][bc]) = bv;
    }
    if (k0 + 32 < Kdim) {   // hint next slab toward the caches
      __builtin_prefetch(Ag + (size_t)(bM + (tid >> 1)) * Kdim + k0 + 32, 0, 1);
      __builtin_prefetch(Bg + (size_t)(k0 + 32 + (tid >> 4)) * Ndim + bN, 0, 1);
    }
    __syncthreads();

    // ---- fragments ----
    v16bf afrag[2];
#pragma unroll
    for (int mi = 0; mi < 2; ++mi) {
      const int Mr = wm * 32 + mi * 16 + ln;
#pragma unroll
      for (int v = 0; v < 8; ++v) {
        const int kv = (v < 4) ? (2 * v + 8 * half) : (16 + 2 * (v - 4) + 8 * half);
        afrag[mi][2 * v]     = As[Mr][kv];
        afrag[mi][2 * v + 1] = As[Mr][kv + 1];
      }
    }
    v16bf bfrag[4];
#pragma unroll
    for (int ni = 0; ni < 4; ++ni) {
      const int Nc = wn * 64 + ni * 16 + ln;
#pragma unroll
      for (int r = 0; r < 8; ++r) {
        const int kr = (r < 4) ? (2 * r + 8 * half) : (16 + 2 * (r - 4) + 8 * half);
        bfrag[ni][2 * r]     = Bs[kr][Nc];
        bfrag[ni][2 * r + 1] = Bs[kr + 1][Nc];
      }
    }
#pragma unroll
    for (int mi = 0; mi < 2; ++mi)
#pragma unroll
      for (int ni = 0; ni < 4; ++ni)
        acc[mi][ni] = __builtin_amdgcn_wmma_f32_16x16x32_bf16(
            false, afrag[mi], false, bfrag[ni], (short)0, acc[mi][ni], false, false);
    __syncthreads();
  }

  // ---- epilogue ----
#pragma unroll
  for (int mi = 0; mi < 2; ++mi)
#pragma unroll
    for (int ni = 0; ni < 4; ++ni)
#pragma unroll
      for (int r = 0; r < 8; ++r) {
        const int grow = bM + wm * 32 + mi * 16 + r + 8 * half;
        const int gcol = bN + wn * 64 + ni * 16 + ln;
        const size_t gi = (size_t)grow * Ndim + gcol;
        const float v = acc[mi][ni][r];
        if (EPI == 0) {
          outF[gi] = v * SCALE_ + ((((gcol >> 4) & 15) == (gcol & 15)) ? 1.f : 0.f);
        } else if (EPI == 1) {
          outF[gi] = v + res[gi];
        } else {
          const float ge = 0.5f * v * (1.f + erff(v * 0.70710678118f));
          outB[gi] = (__bf16)ge;
        }
      }
}

// ---------------------------------------------------------------------------
// Scan pass 1: one wave per (b,h,chunk). Running prefix P lives in LDS;
// each step does P <- P @ m_t (4x v_wmma_f32_16x16x4_f32) and overwrites
// m_t in place with the local inclusive prefix.  Chunk total -> totals.
// ---------------------------------------------------------------------------
__global__ __launch_bounds__(32)
void scan_local_kernel(float* __restrict__ m, float* __restrict__ totals) {
  const int bid   = blockIdx.x;
  const int chunk = bid % NCH_;
  const int h     = (bid / NCH_) % H_;
  const int b     = bid / (NCH_ * H_);
  const int lane  = threadIdx.x;

  __shared__ float P[256];
  {
    const int half = lane >> 4, n = lane & 15;
#pragma unroll
    for (int r = 0; r < 8; ++r) {
      const int mrow = r + 8 * half;
      P[mrow * 16 + n] = (mrow == n) ? 1.f : 0.f;
    }
  }
  __syncthreads();

  for (int t = 0; t < LC_; ++t) {
    const int s = chunk * LC_ + t;
    float* mt = m + (((size_t)b * S_ + s) * H_ + h) * 256;
    v8f d = mat16_mul(P, mt);
    __syncthreads();
    store_d16(P, d);
    store_d16(mt, d);           // in-place: m_t becomes local prefix P_local[t]
    __syncthreads();
  }
  float* tp = totals + (((size_t)b * H_ + h) * NCH_ + chunk) * 256;
  for (int i = lane; i < 256; i += 32) tp[i] = P[i];
}

// ---------------------------------------------------------------------------
// Scan pass 2: one wave per (b,h).  carry[0] = last_state;
// carry[j+1] = carry[j] @ total[j].  Only 32 chunk-steps per wave.
// ---------------------------------------------------------------------------
__global__ __launch_bounds__(32)
void scan_carry_kernel(const float* __restrict__ last_state,
                       const float* __restrict__ totals,
                       float* __restrict__ carries) {
  const int b = blockIdx.x / H_;
  const int h = blockIdx.x % H_;
  const int lane = threadIdx.x;

  __shared__ float P[256];
  const float* ls = last_state + ((size_t)b * H_ + h) * 256;
  for (int i = lane; i < 256; i += 32) P[i] = ls[i];
  __syncthreads();

  for (int j = 0; j < NCH_; ++j) {
    float* cp = carries + (((size_t)b * H_ + h) * NCH_ + j) * 256;
    for (int i = lane; i < 256; i += 32) cp[i] = P[i];   // exclusive carry
    const float* tp = totals + (((size_t)b * H_ + h) * NCH_ + j) * 256;
    v8f d = mat16_mul(P, tp);
    __syncthreads();
    store_d16(P, d);
    __syncthreads();
  }
}

// ---------------------------------------------------------------------------
// Scan pass 3 (fused fixup + head down-projection):
//   state  = carry[chunk] @ P_local[t]           (fp32 WMMA)
//   y_tile = state @ w_down[h] -> bf16 y[B,S,E]  (fp32 WMMA)
//   b == B-1 also writes states[-1] slice of d_out.
// 8 independent waves per block; each wave handles one (b,s,h).
// ---------------------------------------------------------------------------
__global__ __launch_bounds__(256)
void fixup_down_kernel(const float* __restrict__ plocal,
                       const float* __restrict__ carries,
                       const float* __restrict__ w_down,
                       __bf16* __restrict__ y,
                       float* __restrict__ states_out) {
  const int wave = threadIdx.x >> 5;
  const int lane = threadIdx.x & 31;
  const int idx  = blockIdx.x * 8 + wave;     // (b*S + s)*H + h
  const int h = idx % H_;
  const int s = (idx / H_) % S_;
  const int b = idx / (H_ * S_);
  const int chunk = s / LC_;

  const float* carry = carries + (((size_t)b * H_ + h) * NCH_ + chunk) * 256;
  const float* pl    = plocal + (((size_t)b * S_ + s) * H_ + h) * 256;

  v8f st = mat16_mul(carry, pl);              // state matrix for token s

  __shared__ float Sh[8][256];
  store_d16(Sh[wave], st);
  __syncthreads();

  v8f yv = mat16_mul(Sh[wave], w_down + (size_t)h * 256);

  const int half = lane >> 4, c = lane & 15;
#pragma unroll
  for (int r = 0; r < 8; ++r) {
    const int o = r + 8 * half;
    y[((size_t)b * S_ + s) * E_ + h * 256 + o * 16 + c] = (__bf16)yv[r];
    if (b == B_ - 1)
      states_out[((size_t)s * H_ + h) * 256 + o * 16 + c] = st[r];
  }
}

// ---------------------------------------------------------------------------
// host-side orchestration
// ---------------------------------------------------------------------------
extern "C" void kernel_launch(void* const* d_in, const int* in_sizes, int n_in,
                              void* d_out, int out_size, void* d_ws, size_t ws_size,
                              hipStream_t stream) {
  (void)in_sizes; (void)n_in; (void)out_size; (void)ws_size;
  const float* activations = (const float*)d_in[0];   // [B,S,E]
  const float* last_state  = (const float*)d_in[1];   // [B,H,O,O]
  const float* w_up        = (const float*)d_in[2];   // [E, H*O*O] = [2048,2048]
  const float* w_down      = (const float*)d_in[3];   // [H,O,C]
  const float* w_out       = (const float*)d_in[4];   // [E,E]
  const float* ln1_g       = (const float*)d_in[5];
  const float* ln2_g       = (const float*)d_in[6];
  const float* w_mlp1      = (const float*)d_in[7];   // [E,4E]
  const float* w_mlp2      = (const float*)d_in[8];   // [4E,E]
  float* out  = (float*)d_out;                        // act [M,E] ++ states[-1]
  float* states_out = out + (size_t)M_ * E_;

  // ---- workspace layout (bytes); tail region is overlaid ----
  char* ws = (char*)d_ws;
  size_t off = 0;
  auto take = [&](size_t bytes) { size_t r = off; off += (bytes + 255) & ~(size_t)255; return r; };
  const size_t o_wup   = take((size_t)E_ * E_ * 2);
  const size_t o_wout  = take((size_t)E_ * E_ * 2);
  const size_t o_wm1   = take((size_t)E_ * FF_ * 2);
  const size_t o_wm2   = take((size_t)FF_ * E_ * 2);
  const size_t o_xz    = take((size_t)M_ * E_ * 2);        // ln1 out, reused for ln2 out
  const size_t o_dyn   = off;                              // overlay base
  const size_t o_m     = o_dyn;                            // [M,2048] fp32 (new_m -> P_local)
  const size_t o_tot   = o_m   + (size_t)M_ * E_ * 4;
  const size_t o_car   = o_tot + (size_t)B_ * H_ * NCH_ * 256 * 4;
  const size_t o_y     = o_car + (size_t)B_ * H_ * NCH_ * 256 * 4;
  const size_t o_h1    = o_dyn;                            // [M,FF] bf16, overlays m/tot/car/y

  __bf16* wup_b  = (__bf16*)(ws + o_wup);
  __bf16* wout_b = (__bf16*)(ws + o_wout);
  __bf16* wm1_b  = (__bf16*)(ws + o_wm1);
  __bf16* wm2_b  = (__bf16*)(ws + o_wm2);
  __bf16* xz_b   = (__bf16*)(ws + o_xz);
  float*  mbuf   = (float*) (ws + o_m);
  float*  totals = (float*) (ws + o_tot);
  float*  carr   = (float*) (ws + o_car);
  __bf16* y_b    = (__bf16*)(ws + o_y);
  __bf16* h1_b   = (__bf16*)(ws + o_h1);

  // 1) weights -> bf16
  cvt_bf16_kernel<<<(E_ * E_ + 255) / 256, 256, 0, stream>>>(w_up,   wup_b,  E_ * E_);
  cvt_bf16_kernel<<<(E_ * E_ + 255) / 256, 256, 0, stream>>>(w_out,  wout_b, E_ * E_);
  cvt_bf16_kernel<<<(E_ * FF_ + 255) / 256, 256, 0, stream>>>(w_mlp1, wm1_b, E_ * FF_);
  cvt_bf16_kernel<<<(FF_ * E_ + 255) / 256, 256, 0, stream>>>(w_mlp2, wm2_b, FF_ * E_);

  // 2) LN1 -> bf16 x
  ln_bf16_kernel<<<M_, 256, 0, stream>>>(activations, ln1_g, xz_b);

  // 3) new_m = (x @ w_up)*SCALE + I   (fp32 out, [M,2048])
  gemm_bf16_kernel<0><<<dim3(E_ / 128, M_ / 128), 256, 0, stream>>>(
      xz_b, wup_b, mbuf, nullptr, nullptr, E_, E_);

  // 4) local prefix products per 64-token chunk (in place) + chunk totals
  scan_local_kernel<<<B_ * H_ * NCH_, 32, 0, stream>>>(mbuf, totals);

  // 5) exclusive scan of chunk totals seeded with last_state
  scan_carry_kernel<<<B_ * H_, 32, 0, stream>>>(last_state, totals, carr);

  // 6) state fixup + down-projection -> y (bf16) and states[-1] output
  fixup_down_kernel<<<(B_ * S_ * H_) / 8, 256, 0, stream>>>(
      mbuf, carr, w_down, y_b, states_out);

  // 7) act = activations + y @ w_out   (fp32 -> d_out)
  gemm_bf16_kernel<1><<<dim3(E_ / 128, M_ / 128), 256, 0, stream>>>(
      y_b, wout_b, out, nullptr, activations, E_, E_);

  // 8) LN2 -> bf16 z (reuses x buffer)
  ln_bf16_kernel<<<M_, 256, 0, stream>>>(out, ln2_g, xz_b);

  // 9) h1 = gelu(z @ w_mlp1)  (bf16, overlays dead scan buffers)
  gemm_bf16_kernel<2><<<dim3(FF_ / 128, M_ / 128), 256, 0, stream>>>(
      xz_b, wm1_b, nullptr, h1_b, nullptr, FF_, E_);

  // 10) act += h1 @ w_mlp2
  gemm_bf16_kernel<1><<<dim3(E_ / 128, M_ / 128), 256, 0, stream>>>(
      h1_b, wm2_b, out, nullptr, out, E_, FF_);
}
